// SVDNet_29875792511018
// MI455X (gfx1250) — compile-verified, matrix operand order checked by reading, not run on hardware
//
#include <hip/hip_runtime.h>
#include <hip/hip_bf16.h>
#include <cstdint>
#include <math.h>

// ---------------------------------------------------------------------------
// Types for CDNA5 WMMA (wave32): 16x16x32 bf16 -> f32
// ---------------------------------------------------------------------------
typedef __attribute__((ext_vector_type(16))) __bf16 v16bf;
typedef __attribute__((ext_vector_type(8)))  float  v8f;

union FragAB {
    v16bf v;
    uint4 q[2];   // 32 bytes = 16 bf16
};

__device__ __forceinline__ uint16_t f2bf(float f) {
    unsigned u = __float_as_uint(f);
    unsigned r = u + 0x7FFFu + ((u >> 16) & 1u);   // round-to-nearest-even
    return (uint16_t)(r >> 16);
}
__device__ __forceinline__ float bf2f(uint16_t h) {
    return __uint_as_float(((unsigned)h) << 16);
}

// CDNA5 async global->LDS copy (ASYNCcnt-tracked). 16B granule.
__device__ __forceinline__ void async_b128(uint16_t* lds_dst, const uint16_t* gsrc) {
    unsigned l = (unsigned)(uintptr_t)lds_dst;   // low 32 bits = LDS byte offset
    asm volatile("global_load_async_to_lds_b128 %0, %1, off"
                 :: "v"(l), "v"(gsrc) : "memory");
}
__device__ __forceinline__ void wait_async() {
    asm volatile("s_wait_asynccnt 0" ::: "memory");
}

// Padded activation geometry: [C][B=64][H+2=66][W+2=34]
#define PAD_HW   2244          // 66*34
#define CH_STR   143616        // 64*2244
#define P_TOT    131072        // 64*64*32

// ---------------------------------------------------------------------------
// Elementwise f32 -> bf16
// ---------------------------------------------------------------------------
__global__ void cvt_f32_bf16(const float* __restrict__ src, uint16_t* __restrict__ dst, int n) {
    int i = blockIdx.x * 256 + threadIdx.x;
    if (i < n) dst[i] = f2bf(src[i]);
}

// ---------------------------------------------------------------------------
// conv1: 2 -> 64 channels, 3x3 SAME, direct f32 (K=18, trivial).
// x: (B,H,W,2).  Output act1 padded bf16 [64][B][66][34]
// ---------------------------------------------------------------------------
__global__ void conv1_direct(const float* __restrict__ x, const float* __restrict__ w1,
                             const float* __restrict__ b1, uint16_t* __restrict__ act1) {
    int p  = blockIdx.x * 256 + threadIdx.x;   // 0..131071
    int co = blockIdx.y;                       // 0..63
    int b = p >> 11, hw = p & 2047, h = hw >> 5, w = hw & 31;
    float acc = b1[co];
    const float* wp = w1 + co * 18;
    for (int ci = 0; ci < 2; ci++)
        for (int kh = 0; kh < 3; kh++) {
            int hh = h + kh - 1;
            if (hh < 0 || hh >= 64) continue;
            for (int kw = 0; kw < 3; kw++) {
                int ww = w + kw - 1;
                if (ww < 0 || ww >= 32) continue;
                acc += x[((b * 64 + hh) * 32 + ww) * 2 + ci] * wp[ci * 9 + kh * 3 + kw];
            }
        }
    act1[co * CH_STR + b * PAD_HW + (h + 1) * 34 + (w + 1)] = f2bf(fmaxf(acc, 0.f));
}

// ---------------------------------------------------------------------------
// Implicit-GEMM conv (3x3 SAME) with WMMA bf16, padded input (no bounds checks).
// Block: 256 threads (8 waves). Block tile 64 Co x 128 px; wave tile 32x32
// -> 4 WMMAs per K-step per wave. K-step = 32.
// A tile staged with CDNA5 async global->LDS b128; B tile im2col-gathered with
// an incremental (ci,kh,kw) walk and packed ds_store_b128.
// outPadded=1 -> write padded layout (feeds next conv); 0 -> flat [Co][P].
// ---------------------------------------------------------------------------
__global__ __launch_bounds__(256) void conv_igemm(
        const uint16_t* __restrict__ actIn, const uint16_t* __restrict__ wB,
        const float* __restrict__ bias, uint16_t* __restrict__ actOut,
        int Ci, int outPadded) {
    const int K = Ci * 9;
    __shared__ __align__(16) uint16_t lA[64 * 32];    // 4 KB
    __shared__ __align__(16) uint16_t lB[128 * 32];   // 8 KB

    int tid   = threadIdx.x;
    int pBlk  = blockIdx.x * 128;
    int coBlk = blockIdx.y * 64;
    int wid = tid >> 5, lane = tid & 31;
    int co2 = wid >> 2, px4 = wid & 3;
    int khalf = lane >> 4, l15 = lane & 15;

    v8f c00 = {}, c01 = {}, c10 = {}, c11 = {};

    int aRow  = co2 * 32 + l15;        // fragment rows in lA
    int bRow0 = px4 * 32 + l15;        // fragment rows in lB
    // A staging: 64 rows x 32 cols, 1 async b128 per thread
    int asRow = tid >> 2, asCol = (tid & 3) * 8;
    // B staging: 128 px x 32 k, 16 elements (one half-row) per thread
    int px      = tid >> 1;
    int colBase = (tid & 1) * 16;
    int p  = pBlk + px;
    int pb = p >> 11, ph = (p >> 5) & 63, pw = p & 31;
    int pixBase = pb * PAD_HW + ph * 34 + pw;   // +kh*34+kw indexes padded halo

    for (int k0 = 0; k0 < K; k0 += 32) {
        // ---- A tile: async global -> LDS ----
        async_b128(lA + asRow * 32 + asCol, wB + (coBlk + asRow) * K + k0 + asCol);

        // ---- B tile: im2col gather (no bounds checks; padded input) ----
        {
            int kc = k0 + colBase;
            int ci = kc / 9;                       // one division per K-step
            int r  = kc - ci * 9;
            int kh = (r >= 6) ? 2 : ((r >= 3) ? 1 : 0);
            int kw = r - kh * 3;
            union { uint4 q[2]; uint16_t u[16]; } tmp;
            const uint16_t* src = actIn + pixBase;
#pragma unroll
            for (int i = 0; i < 16; i++) {
                tmp.u[i] = src[ci * CH_STR + kh * 34 + kw];
                kw++;
                if (kw == 3) { kw = 0; kh++; if (kh == 3) { kh = 0; ci++; } }
            }
            *(uint4*)(lB + px * 32 + colBase)     = tmp.q[0];
            *(uint4*)(lB + px * 32 + colBase + 8) = tmp.q[1];
        }
        wait_async();
        __syncthreads();

        FragAB a0, a1, b0, b1;
        a0.q[0] = *(const uint4*)(lA + aRow * 32 + khalf * 8);
        a0.q[1] = *(const uint4*)(lA + aRow * 32 + 16 + khalf * 8);
        a1.q[0] = *(const uint4*)(lA + (aRow + 16) * 32 + khalf * 8);
        a1.q[1] = *(const uint4*)(lA + (aRow + 16) * 32 + 16 + khalf * 8);
        b0.q[0] = *(const uint4*)(lB + bRow0 * 32 + khalf * 8);
        b0.q[1] = *(const uint4*)(lB + bRow0 * 32 + 16 + khalf * 8);
        b1.q[0] = *(const uint4*)(lB + (bRow0 + 16) * 32 + khalf * 8);
        b1.q[1] = *(const uint4*)(lB + (bRow0 + 16) * 32 + 16 + khalf * 8);

        c00 = __builtin_amdgcn_wmma_f32_16x16x32_bf16(false, a0.v, false, b0.v, (short)0, c00, false, false);
        c01 = __builtin_amdgcn_wmma_f32_16x16x32_bf16(false, a0.v, false, b1.v, (short)0, c01, false, false);
        c10 = __builtin_amdgcn_wmma_f32_16x16x32_bf16(false, a1.v, false, b0.v, (short)0, c10, false, false);
        c11 = __builtin_amdgcn_wmma_f32_16x16x32_bf16(false, a1.v, false, b1.v, (short)0, c11, false, false);
        __syncthreads();
    }

    // Epilogue: bias + relu + bf16 store (fully unrolled: no movrels)
    int hi8 = (lane & 16) ? 8 : 0;
    int n0 = pBlk + px4 * 32 + l15;
    int n1 = n0 + 16;
    int mBase = coBlk + co2 * 32 + hi8;
    if (outPadded) {
        int pad0 = (n0 >> 11) * PAD_HW + (((n0 >> 5) & 63) + 1) * 34 + (n0 & 31) + 1;
        int pad1 = (n1 >> 11) * PAD_HW + (((n1 >> 5) & 63) + 1) * 34 + (n1 & 31) + 1;
#pragma unroll
        for (int r = 0; r < 8; r++) {
            int m0 = mBase + r, m1 = m0 + 16;
            float bi0 = bias[m0], bi1 = bias[m1];
            actOut[m0 * CH_STR + pad0] = f2bf(fmaxf(c00[r] + bi0, 0.f));
            actOut[m0 * CH_STR + pad1] = f2bf(fmaxf(c01[r] + bi0, 0.f));
            actOut[m1 * CH_STR + pad0] = f2bf(fmaxf(c10[r] + bi1, 0.f));
            actOut[m1 * CH_STR + pad1] = f2bf(fmaxf(c11[r] + bi1, 0.f));
        }
    } else {
#pragma unroll
        for (int r = 0; r < 8; r++) {
            int m0 = mBase + r, m1 = m0 + 16;
            float bi0 = bias[m0], bi1 = bias[m1];
            actOut[m0 * P_TOT + n0] = f2bf(fmaxf(c00[r] + bi0, 0.f));
            actOut[m0 * P_TOT + n1] = f2bf(fmaxf(c01[r] + bi0, 0.f));
            actOut[m1 * P_TOT + n0] = f2bf(fmaxf(c10[r] + bi1, 0.f));
            actOut[m1 * P_TOT + n1] = f2bf(fmaxf(c11[r] + bi1, 0.f));
        }
    }
}

// ---------------------------------------------------------------------------
// 4x4 average pool over (16,8) blocks, act3 flat [256][B*2048] -> fpool bf16
// [B][4096]; feature = c*16 + ph*4 + pw (matches reshape(B,256,4,4).reshape(B,-1))
// ---------------------------------------------------------------------------
__global__ void pool_kernel(const uint16_t* __restrict__ act3, uint16_t* __restrict__ fpool) {
    int idx = blockIdx.x * 256 + threadIdx.x;   // 64*4096
    int b = idx >> 12;
    int feat = idx & 4095;
    int c = feat >> 4, ph = (feat >> 2) & 3, pw = feat & 3;
    const uint16_t* base = act3 + c * P_TOT + b * 2048;
    float s = 0.f;
    for (int h = ph * 16; h < ph * 16 + 16; h++)
#pragma unroll
        for (int w = pw * 8; w < pw * 8 + 8; w++)
            s += bf2f(base[h * 32 + w]);
    fpool[b * 4096 + feat] = f2bf(s * (1.f / 128.f));
}

// ---------------------------------------------------------------------------
// Dense GEMM:  out[n][m] = A[m][:] . Bt[n][:] + bias[m]
// Block tile 128 M x 64 N; wave tile 32x32 -> 4 WMMAs per K-step.
// Both tiles staged with async global->LDS b128.
// mode 0: relu -> bf16 out;  mode 1: f32 out (no relu)
// grid.x = N/64 (here 1), grid.y = M/128
// ---------------------------------------------------------------------------
__global__ __launch_bounds__(256) void gemm_bf16(
        const uint16_t* __restrict__ A, const uint16_t* __restrict__ Bt,
        const float* __restrict__ bias, void* __restrict__ Out,
        int Mtot, int K, int mode) {
    __shared__ __align__(16) uint16_t lA[128 * 32];   // 8 KB
    __shared__ __align__(16) uint16_t lB[64 * 32];    // 4 KB

    int tid  = threadIdx.x;
    int nBlk = blockIdx.x * 64;
    int mBlk = blockIdx.y * 128;
    int wid = tid >> 5, lane = tid & 31;
    int co4 = wid >> 1, px2 = wid & 1;
    int khalf = lane >> 4, l15 = lane & 15;

    v8f c00 = {}, c01 = {}, c10 = {}, c11 = {};
    int aRow  = co4 * 32 + l15;
    int bRow0 = px2 * 32 + l15;
    int asRow = tid >> 1, asCol = (tid & 1) * 16;   // A: 2 async b128 / thread
    int bsRow = tid >> 2, bsCol = (tid & 3) * 8;    // B: 1 async b128 / thread

    for (int k0 = 0; k0 < K; k0 += 32) {
        const uint16_t* aSrc = A + (mBlk + asRow) * K + k0 + asCol;
        async_b128(lA + asRow * 32 + asCol,     aSrc);
        async_b128(lA + asRow * 32 + asCol + 8, aSrc + 8);
        async_b128(lB + bsRow * 32 + bsCol, Bt + (nBlk + bsRow) * K + k0 + bsCol);
        wait_async();
        __syncthreads();

        FragAB a0, a1, b0, b1;
        a0.q[0] = *(const uint4*)(lA + aRow * 32 + khalf * 8);
        a0.q[1] = *(const uint4*)(lA + aRow * 32 + 16 + khalf * 8);
        a1.q[0] = *(const uint4*)(lA + (aRow + 16) * 32 + khalf * 8);
        a1.q[1] = *(const uint4*)(lA + (aRow + 16) * 32 + 16 + khalf * 8);
        b0.q[0] = *(const uint4*)(lB + bRow0 * 32 + khalf * 8);
        b0.q[1] = *(const uint4*)(lB + bRow0 * 32 + 16 + khalf * 8);
        b1.q[0] = *(const uint4*)(lB + (bRow0 + 16) * 32 + khalf * 8);
        b1.q[1] = *(const uint4*)(lB + (bRow0 + 16) * 32 + 16 + khalf * 8);

        c00 = __builtin_amdgcn_wmma_f32_16x16x32_bf16(false, a0.v, false, b0.v, (short)0, c00, false, false);
        c01 = __builtin_amdgcn_wmma_f32_16x16x32_bf16(false, a0.v, false, b1.v, (short)0, c01, false, false);
        c10 = __builtin_amdgcn_wmma_f32_16x16x32_bf16(false, a1.v, false, b0.v, (short)0, c10, false, false);
        c11 = __builtin_amdgcn_wmma_f32_16x16x32_bf16(false, a1.v, false, b1.v, (short)0, c11, false, false);
        __syncthreads();
    }

    int hi8 = (lane & 16) ? 8 : 0;
    int n0 = nBlk + px2 * 32 + l15;
    int n1 = n0 + 16;
    int mBase = mBlk + co4 * 32 + hi8;
    if (mode == 0) {
        uint16_t* o = (uint16_t*)Out;
#pragma unroll
        for (int r = 0; r < 8; r++) {
            int m0 = mBase + r, m1 = m0 + 16;
            float bi0 = bias[m0], bi1 = bias[m1];
            o[n0 * Mtot + m0] = f2bf(fmaxf(c00[r] + bi0, 0.f));
            o[n1 * Mtot + m0] = f2bf(fmaxf(c01[r] + bi0, 0.f));
            o[n0 * Mtot + m1] = f2bf(fmaxf(c10[r] + bi1, 0.f));
            o[n1 * Mtot + m1] = f2bf(fmaxf(c11[r] + bi1, 0.f));
        }
    } else {
        float* o = (float*)Out;
#pragma unroll
        for (int r = 0; r < 8; r++) {
            int m0 = mBase + r, m1 = m0 + 16;
            float bi0 = bias[m0], bi1 = bias[m1];
            o[n0 * Mtot + m0] = c00[r] + bi0;
            o[n1 * Mtot + m0] = c01[r] + bi0;
            o[n0 * Mtot + m1] = c10[r] + bi1;
            o[n1 * Mtot + m1] = c11[r] + bi1;
        }
    }
}

// ---------------------------------------------------------------------------
// sigma head: per sample b, sigma[r] = softplus(f[b].ws[r] + bs[r]), sorted desc
// ---------------------------------------------------------------------------
__global__ void sigma_head(const uint16_t* __restrict__ f, const float* __restrict__ wsw,
                           const float* __restrict__ bs, float* __restrict__ out) {
    int b = blockIdx.x;
    int t = threadIdx.x;
    int r = t >> 5, lane = t & 31;
    float part = 0.f;
    for (int k = lane; k < 2048; k += 32)
        part += bf2f(f[b * 2048 + k]) * wsw[r * 2048 + k];
    __shared__ float red[256];
    __shared__ float sig[8];
    red[t] = part;
    __syncthreads();
    if (lane == 0) {
        float s = 0.f;
        for (int i = 0; i < 32; i++) s += red[r * 32 + i];
        s += bs[r];
        sig[r] = (s > 20.f) ? s : log1pf(expf(s));   // softplus
    }
    __syncthreads();
    if (t == 0) {
        float v[8];
        for (int i = 0; i < 8; i++) v[i] = sig[i];
        for (int i = 1; i < 8; i++) {            // insertion sort descending
            float key = v[i]; int j = i - 1;
            while (j >= 0 && v[j] < key) { v[j + 1] = v[j]; j--; }
            v[j + 1] = key;
        }
        for (int i = 0; i < 8; i++) out[b * 8 + i] = v[i];
    }
}

// ---------------------------------------------------------------------------
// Modified complex Gram-Schmidt (matches reference). One block per sample,
// blockDim = Mdim (64 for u, 32 for v), thread = row. R = 8 columns.
// ---------------------------------------------------------------------------
__global__ void gram_schmidt(const float* __restrict__ raw, float* __restrict__ outp, int Mdim) {
    int b = blockIdx.x;
    int tid = threadIdx.x;
    __shared__ float sa[64], sb[64], sc[64];
    __shared__ float res[3];
    float2 q[8];
    const float* rb = raw + (size_t)b * Mdim * 16;   // Mdim * R * 2

    for (int i = 0; i < 8; i++) {
        float vr = rb[tid * 16 + i * 2];
        float vi = rb[tid * 16 + i * 2 + 1];
        for (int j = 0; j < i; j++) {
            float qr = q[j].x, qi = q[j].y;
            sa[tid] = qr * vr + qi * vi;
            sb[tid] = qr * vi - qi * vr;
            sc[tid] = qr * qr + qi * qi;
            __syncthreads();
            if (tid == 0) {
                float A = 0, B = 0, C = 0;
                for (int m = 0; m < (int)blockDim.x; m++) { A += sa[m]; B += sb[m]; C += sc[m]; }
                res[0] = A; res[1] = B; res[2] = C;
            }
            __syncthreads();
            float tr = res[0] / res[2], ti = res[1] / res[2];
            vr -= tr * q[j].x - ti * q[j].y;
            vi -= tr * q[j].y + ti * q[j].x;
        }
        sa[tid] = vr * vr + vi * vi;
        __syncthreads();
        if (tid == 0) {
            float s = 0;
            for (int m = 0; m < (int)blockDim.x; m++) s += sa[m];
            res[0] = sqrtf(s);
        }
        __syncthreads();
        float nrm = res[0];
        if (nrm > 1e-6f) { vr /= nrm; vi /= nrm; }
        q[i] = make_float2(vr, vi);
        size_t o = ((size_t)b * Mdim * 8 + (size_t)tid * 8 + i) * 2;
        outp[o] = vr;
        outp[o + 1] = vi;
    }
}

// ---------------------------------------------------------------------------
// Launch
// ---------------------------------------------------------------------------
extern "C" void kernel_launch(void* const* d_in, const int* in_sizes, int n_in,
                              void* d_out, int out_size, void* d_ws, size_t ws_size,
                              hipStream_t stream) {
    const float* x   = (const float*)d_in[0];
    const float* w1  = (const float*)d_in[1];
    const float* b1  = (const float*)d_in[2];
    const float* w2  = (const float*)d_in[3];
    const float* b2  = (const float*)d_in[4];
    const float* w3  = (const float*)d_in[5];
    const float* b3  = (const float*)d_in[6];
    const float* wc  = (const float*)d_in[7];
    const float* bc  = (const float*)d_in[8];
    const float* wu  = (const float*)d_in[9];
    const float* bu  = (const float*)d_in[10];
    const float* wv  = (const float*)d_in[11];
    const float* bv  = (const float*)d_in[12];
    const float* wsw = (const float*)d_in[13];
    const float* bs  = (const float*)d_in[14];
    float* out = (float*)d_out;

    char* base = (char*)d_ws;
    size_t off = 0;
    auto alloc = [&](size_t bytes) -> void* {
        void* p = base + off;
        off = (off + bytes + 255) & ~(size_t)255;
        return p;
    };
    size_t act1B = (size_t)64  * CH_STR * 2;   // padded
    size_t act2B = (size_t)128 * CH_STR * 2;   // padded
    uint16_t* act1  = (uint16_t*)alloc(act1B);
    uint16_t* act2  = (uint16_t*)alloc(act2B);
    uint16_t* act3  = (uint16_t*)alloc((size_t)256 * P_TOT * 2);   // flat
    uint16_t* w2b   = (uint16_t*)alloc((size_t)128 * 576 * 2);
    uint16_t* w3b   = (uint16_t*)alloc((size_t)256 * 1152 * 2);
    uint16_t* wcb   = (uint16_t*)alloc((size_t)2048 * 4096 * 2);
    uint16_t* wub   = (uint16_t*)alloc((size_t)1024 * 2048 * 2);
    uint16_t* wvb   = (uint16_t*)alloc((size_t)512 * 2048 * 2);
    uint16_t* fpool = (uint16_t*)alloc((size_t)64 * 4096 * 2);
    uint16_t* fbuf  = (uint16_t*)alloc((size_t)64 * 2048 * 2);
    float*    uraw  = (float*)alloc((size_t)64 * 1024 * 4);
    float*    vraw  = (float*)alloc((size_t)64 * 512 * 4);

    // zero the padded halos (interior is fully rewritten each call)
    hipMemsetAsync(act1, 0, act1B, stream);
    hipMemsetAsync(act2, 0, act2B, stream);

    auto cvt = [&](const float* s, uint16_t* d, int n) {
        cvt_f32_bf16<<<(n + 255) / 256, 256, 0, stream>>>(s, d, n);
    };
    cvt(w2, w2b, 128 * 576);
    cvt(w3, w3b, 256 * 1152);
    cvt(wc, wcb, 2048 * 4096);
    cvt(wu, wub, 1024 * 2048);
    cvt(wv, wvb, 512 * 2048);

    conv1_direct<<<dim3(512, 64), 256, 0, stream>>>(x, w1, b1, act1);
    conv_igemm<<<dim3(1024, 2), 256, 0, stream>>>(act1, w2b, b2, act2, 64, 1);
    conv_igemm<<<dim3(1024, 4), 256, 0, stream>>>(act2, w3b, b3, act3, 128, 0);
    pool_kernel<<<dim3(1024), 256, 0, stream>>>(act3, fpool);
    gemm_bf16<<<dim3(1, 16), 256, 0, stream>>>(wcb, fpool, bc, (void*)fbuf, 2048, 4096, 0);
    gemm_bf16<<<dim3(1, 8), 256, 0, stream>>>(wub, fbuf, bu, (void*)uraw, 1024, 2048, 1);
    gemm_bf16<<<dim3(1, 4), 256, 0, stream>>>(wvb, fbuf, bv, (void*)vraw, 512, 2048, 1);

    // output layout: u (64*64*8 complex = 65536 f32) | sigma (512) | v (32768)
    sigma_head<<<dim3(64), 256, 0, stream>>>(fbuf, wsw, bs, out + 65536);
    gram_schmidt<<<dim3(64), 64, 0, stream>>>(uraw, out, 64);
    gram_schmidt<<<dim3(64), 32, 0, stream>>>(vraw, out + 66048, 32);
}
